// TestModel_20864951124663
// MI455X (gfx1250) — compile-verified, matrix operand order checked by reading, not run on hardware
//
#include <hip/hip_runtime.h>
#include <hip/hip_bf16.h>

#define FEAT 128
#define HIDDEN 128
#define NCLS 40
#define WPAD (FEAT + 8)   // bf16 row stride 136 (272 B): 16B-aligned, bank-conflict-free

typedef __attribute__((ext_vector_type(16))) __bf16 v16bf;
typedef __attribute__((ext_vector_type(8)))  float  v8f;
typedef __attribute__((ext_vector_type(4)))  unsigned int u32x4;
typedef __attribute__((ext_vector_type(8)))  int i32x8;
typedef __attribute__((ext_vector_type(4)))  int i32x4;

__device__ __forceinline__ unsigned short f2bf(float f) {
    union { float f; unsigned int u; } a; a.f = f;
    unsigned int u = a.u + 0x7FFFu + ((a.u >> 16) & 1u);   // round-to-nearest-even
    return (unsigned short)(u >> 16);
}

union ABFrag { v16bf v; unsigned short u[16]; uint4 q[2]; };

// ---------------------------------------------------------------- init
__global__ void gcn_init(float* __restrict__ deg, float* __restrict__ acc, int N) {
    long long i = (long long)blockIdx.x * blockDim.x + threadIdx.x;
    long long tot = (long long)N * FEAT;
    if (i < tot) acc[i] = 0.0f;
    if (i < N)   deg[i] = 1.0f;          // self-loop
}

// ---------------------------------------------------------------- degree count
__global__ void gcn_edge_deg(const long long* __restrict__ ei, float* __restrict__ deg,
                             long long E) {
    long long i = (long long)blockIdx.x * blockDim.x + threadIdx.x;
    if (i < E) unsafeAtomicAdd(&deg[ei[E + i]], 1.0f);     // col = destination
}

// ---------------------------------------------------------------- deg -> deg^{-1/2}
__global__ void gcn_rsqrt(float* __restrict__ deg, int N) {
    long long i = (long long)blockIdx.x * blockDim.x + threadIdx.x;
    if (i < N) {
        float d = deg[i];
        deg[i] = (d > 0.0f) ? rsqrtf(d) : 0.0f;
    }
}

// ---------------------------------------------------------------- GEMM1: h = x @ W1 (bf16 WMMA)
// block = 256 threads (8 waves); block covers 16 rows x 128 cols, one 16x16 tile/wave.
// W1 staged into LDS by the Tensor Data Mover, then converted to transposed bf16.
__global__ void gcn_gemm1(const float* __restrict__ x, const float* __restrict__ W,
                          float* __restrict__ h, int N) {
    __shared__ float sWf[FEAT * HIDDEN];                    // 64 KB TDM landing pad
    __shared__ unsigned short sWT[HIDDEN][WPAD];            // bf16, [n][k], padded

#if __has_builtin(__builtin_amdgcn_tensor_load_to_lds)
    if (threadIdx.x == 0) {
        unsigned long long ga = (unsigned long long)(const void*)W;
        unsigned ldsOff = (unsigned)(size_t)&sWf[0];        // low 32 bits = LDS offset
        // D# group0: count=1 | lds_addr | global_addr[56:0] | type=2 (bits 127:126)
        u32x4 g0 = { 1u, ldsOff,
                     (unsigned)(ga & 0xFFFFFFFFull),
                     (unsigned)((ga >> 32) & 0x01FFFFFFull) | 0x80000000u };
        // D# group1: data_size=4B; tensor_dim0=128, tensor_dim1=128;
        //            tile_dim0=128, tile_dim1=128; tensor_dim0_stride=128
        i32x8 g1 = { 0x00020000,            // wg_mask=0 | data_size=2 (4B)
                     (int)(128u << 16),     // [63:48] tensor_dim0 lo16
                     (int)(128u << 16),     // [79:64] dim0 hi=0 | [95:80] tensor_dim1 lo16
                     (int)(128u << 16),     // [111:96] dim1 hi=0 | [127:112] tile_dim0
                     128,                   // [143:128] tile_dim1 | [159:144] tile_dim2=0
                     128,                   // tensor_dim0_stride lo32
                     0, 0 };
        i32x4 gz4 = { 0, 0, 0, 0 };
#if defined(__clang_major__) && (__clang_major__ >= 23)
        i32x8 gz8 = { 0, 0, 0, 0, 0, 0, 0, 0 };
        __builtin_amdgcn_tensor_load_to_lds(g0, g1, gz4, gz4, gz8, 0);
#else
        __builtin_amdgcn_tensor_load_to_lds(g0, g1, gz4, gz4, 0);
#endif
    }
    __builtin_amdgcn_s_wait_tensorcnt(0);
    __syncthreads();
#else
    for (int i = threadIdx.x; i < FEAT * HIDDEN; i += 256) sWf[i] = W[i];
    __syncthreads();
#endif

    // convert + transpose: sWT[n][k] = bf16(W[k][n])
    for (int i = threadIdx.x; i < FEAT * HIDDEN; i += 256) {
        int k = i >> 7, n = i & 127;
        sWT[n][k] = f2bf(sWf[i]);
    }
    __syncthreads();

    const int wave = threadIdx.x >> 5;
    const int lane = threadIdx.x & 31;
    const int m    = lane & 15;                            // A row / B col within tile
    const int hi   = lane >> 4;                            // K-half select (ISA 7.12.2)
    const int rowBase = blockIdx.x * 16;
    const int n0      = wave * 16;

    int row = rowBase + m; if (row >= N) row = N - 1;      // clamp: keep EXEC all-1s
    const float* xr = x + (size_t)row * FEAT;
    const unsigned short* bp = &sWT[n0 + m][0];

    v8f acc = {};
    #pragma unroll
    for (int kk = 0; kk < FEAT; kk += 32) {
        const int kb = kk + hi * 8;
        ABFrag a, b;
        // A: rows of x, 4x float4 (global_load_b128 each)
        float4 p0 = *(const float4*)(xr + kb);
        float4 p1 = *(const float4*)(xr + kb + 4);
        float4 p2 = *(const float4*)(xr + kb + 16);
        float4 p3 = *(const float4*)(xr + kb + 20);
        a.u[0]  = f2bf(p0.x); a.u[1]  = f2bf(p0.y); a.u[2]  = f2bf(p0.z); a.u[3]  = f2bf(p0.w);
        a.u[4]  = f2bf(p1.x); a.u[5]  = f2bf(p1.y); a.u[6]  = f2bf(p1.z); a.u[7]  = f2bf(p1.w);
        a.u[8]  = f2bf(p2.x); a.u[9]  = f2bf(p2.y); a.u[10] = f2bf(p2.z); a.u[11] = f2bf(p2.w);
        a.u[12] = f2bf(p3.x); a.u[13] = f2bf(p3.y); a.u[14] = f2bf(p3.z); a.u[15] = f2bf(p3.w);
        // B: contiguous bf16 runs from transposed LDS (2x ds_load_b128)
        b.q[0] = *(const uint4*)(bp + kb);
        b.q[1] = *(const uint4*)(bp + kb + 16);
        acc = __builtin_amdgcn_wmma_f32_16x16x32_bf16(
            /*neg_a=*/false, a.v, /*neg_b=*/false, b.v,
            /*c_mod=*/(short)0, acc, /*reuse_a=*/false, /*reuse_b=*/false);
    }

    // C/D layout: lanes 0-15 -> N=lane, M=r; lanes 16-31 -> N=lane-16, M=r+8
    float* hp = h + (size_t)rowBase * HIDDEN + n0 + m;
    if (rowBase + 16 <= N) {                               // uniform fast path
        #pragma unroll
        for (int r = 0; r < 8; ++r)
            hp[(size_t)(r + hi * 8) * HIDDEN] = acc[r];
    } else {
        #pragma unroll
        for (int r = 0; r < 8; ++r) {
            int orow = rowBase + r + hi * 8;
            if (orow < N) h[(size_t)orow * HIDDEN + n0 + m] = acc[r];
        }
    }
}

// ---------------------------------------------------------------- edge scatter-add
// one wave per edge; each lane moves float4 (128 floats / 32 lanes)
__global__ void gcn_scatter(const long long* __restrict__ ei, const float* __restrict__ h,
                            const float* __restrict__ dinv, float* __restrict__ acc,
                            long long E) {
    long long gid = (long long)blockIdx.x * blockDim.x + threadIdx.x;
    long long e   = gid >> 5;
    int lane      = threadIdx.x & 31;
    if (e >= E) return;

    long long r = ei[e];
    long long c = ei[E + e];
    __builtin_prefetch(h + (size_t)r * HIDDEN + lane * 4, 0, 1);  // global_prefetch_b8
    float norm = dinv[r] * dinv[c];

    float4 v = ((const float4*)(h + (size_t)r * HIDDEN))[lane];
    float* out = acc + (size_t)c * HIDDEN + lane * 4;
    unsafeAtomicAdd(out + 0, norm * v.x);
    unsafeAtomicAdd(out + 1, norm * v.y);
    unsafeAtomicAdd(out + 2, norm * v.z);
    unsafeAtomicAdd(out + 3, norm * v.w);
}

// ---------------------------------------------------------------- finalize:
// out = log_softmax( relu(acc + dinv^2*h + b1) @ lin_W + lin_b )
__global__ void gcn_finalize(const float* __restrict__ acc, const float* __restrict__ h,
                             const float* __restrict__ dinv, const float* __restrict__ b1,
                             const float* __restrict__ linW, const float* __restrict__ linb,
                             float* __restrict__ out, int N) {
    __shared__ __align__(16) float sv[8][HIDDEN];
    const int wave = threadIdx.x >> 5;
    const int lane = threadIdx.x & 31;
    const int node = blockIdx.x * 8 + wave;
    const bool active = node < N;

    if (active) {
        float s = dinv[node]; s *= s;                      // self-loop norm
        float4 a  = ((const float4*)(acc + (size_t)node * HIDDEN))[lane];
        float4 hv = ((const float4*)(h   + (size_t)node * HIDDEN))[lane];
        float4 bb = ((const float4*)b1)[lane];
        float4 v;
        v.x = fmaxf(a.x + s * hv.x + bb.x, 0.0f);
        v.y = fmaxf(a.y + s * hv.y + bb.y, 0.0f);
        v.z = fmaxf(a.z + s * hv.z + bb.z, 0.0f);
        v.w = fmaxf(a.w + s * hv.w + bb.w, 0.0f);
        ((float4*)&sv[wave][0])[lane] = v;
    }
    __syncthreads();

    if (active) {
        const int c1 = (lane < 8) ? (32 + lane) : (NCLS - 1);   // clamped, masked later
        float acc0 = linb[lane];
        float acc1 = linb[c1];
        #pragma unroll 4
        for (int f = 0; f < HIDDEN; ++f) {
            float vf = sv[wave][f];
            acc0 = fmaf(vf, linW[f * NCLS + lane], acc0);
            acc1 = fmaf(vf, linW[f * NCLS + c1],   acc1);
        }
        float vmax = fmaxf(acc0, (lane < 8) ? acc1 : -3.402823466e38f);
        for (int off = 16; off > 0; off >>= 1)
            vmax = fmaxf(vmax, __shfl_xor(vmax, off, 32));
        float ssum = __expf(acc0 - vmax) + ((lane < 8) ? __expf(acc1 - vmax) : 0.0f);
        for (int off = 16; off > 0; off >>= 1)
            ssum += __shfl_xor(ssum, off, 32);
        float lse = vmax + __logf(ssum);

        float* op = out + (size_t)node * NCLS;
        op[lane] = acc0 - lse;
        if (lane < 8) op[32 + lane] = acc1 - lse;
    }
}

// ---------------------------------------------------------------- launch
extern "C" void kernel_launch(void* const* d_in, const int* in_sizes, int n_in,
                              void* d_out, int out_size, void* d_ws, size_t ws_size,
                              hipStream_t stream) {
    const float*     x    = (const float*)d_in[0];
    const long long* ei   = (const long long*)d_in[1];    // int64 [2, E]
    const float*     W1   = (const float*)d_in[2];
    const float*     b1   = (const float*)d_in[3];
    const float*     linW = (const float*)d_in[4];
    const float*     linb = (const float*)d_in[5];
    float*           out  = (float*)d_out;

    const int       N = in_sizes[0] / FEAT;
    const long long E = (long long)in_sizes[1] / 2;

    float* ws_h   = (float*)d_ws;                         // N*128
    float* ws_acc = ws_h + (size_t)N * HIDDEN;            // N*128
    float* ws_deg = ws_acc + (size_t)N * HIDDEN;          // N (becomes deg^{-1/2})

    const long long initTot = (long long)N * HIDDEN;
    gcn_init<<<(unsigned)((initTot + 255) / 256), 256, 0, stream>>>(ws_deg, ws_acc, N);
    gcn_edge_deg<<<(unsigned)((E + 255) / 256), 256, 0, stream>>>(ei, ws_deg, E);
    gcn_rsqrt<<<(unsigned)((N + 255) / 256), 256, 0, stream>>>(ws_deg, N);
    gcn_gemm1<<<(unsigned)((N + 15) / 16), 256, 0, stream>>>(x, W1, ws_h, N);
    gcn_scatter<<<(unsigned)((E * 32 + 255) / 256), 256, 0, stream>>>(ei, ws_h, ws_deg, ws_acc, E);
    gcn_finalize<<<(unsigned)((N + 7) / 8), 256, 0, stream>>>(ws_acc, ws_h, ws_deg, b1,
                                                              linW, linb, out, N);
}
